// Llama4TextExperts_42777874268733
// MI455X (gfx1250) — matrix operand order repaired and checked by reading, not compiled, not run on hardware
//
#include <hip/hip_runtime.h>

// Llama4 MoE experts: E=8, H=2048, D=4096, T=2048 tokens/expert.
// Phase 1: fused = up * silu(gate)  (bf16 WMMA, SwiGLU epilogue, bf16 -> ws)
// Phase 2: out = fused @ down_proj  (bf16 WMMA, fp32 out)
// Ping-pong LDS (1 barrier / K-step), hardware bf16 converts, rcp-based sigmoid.

#define E_ 8
#define H_ 2048
#define D_ 4096
#define T_ 2048
#define KP 40  // LDS K stride in bf16 elems (80 B = 5*16 B -> b128-aligned, bank-spread)

typedef __attribute__((ext_vector_type(16))) __bf16 v16bf;
typedef __attribute__((ext_vector_type(2)))  __bf16 v2bf;
typedef __attribute__((ext_vector_type(8)))  float  v8f;
typedef unsigned short u16;
typedef unsigned int   u32;

union FragAB { v16bf v; uint4 u[2]; };

__device__ __forceinline__ u32 pkbf(float x, float y) {  // 2xf32 -> packed bf16
    v2bf v; v[0] = (__bf16)x; v[1] = (__bf16)y;
    return __builtin_bit_cast(u32, v);
}

// ---------------------------------------------------------------------------
// Kernel 1: gate & up GEMMs fused + SwiGLU, writes bf16 activation to ws.
// Block tile 128(T) x 64(D), K-step 32 over H. 8 waves, 32x32 each.
// ---------------------------------------------------------------------------
__global__ __launch_bounds__(256, 2)
void gateup_kernel(const float* __restrict__ hs,
                   const float* __restrict__ Wg,
                   const float* __restrict__ Wu,
                   u16* __restrict__ fused)
{
    __shared__ __align__(16) u16 ldsA[2][128 * KP];  // [M][K]
    __shared__ __align__(16) u16 ldsG[2][64 * KP];   // [N][K] (transposed)
    __shared__ __align__(16) u16 ldsU[2][64 * KP];

    const int tid  = threadIdx.x;
    const int lane = tid & 31;
    const int wv   = tid >> 5;
    const int e    = blockIdx.z;
    const int m0   = blockIdx.y * 128;
    const int n0   = blockIdx.x * 64;

    const float* Ab = hs + (size_t)(e * T_ + m0) * H_;
    const float* Gb = Wg + (size_t)e * H_ * D_ + n0;
    const float* Ub = Wu + (size_t)e * H_ * D_ + n0;

    const int wm = (wv >> 1) * 32;
    const int wn = (wv & 1)  * 32;
    const int mr = lane & 15;
    const int hf = lane >> 4;

    // loader index precompute
    const int rA  = tid >> 2;          // A row (0..63), +64 for q=1
    const int kA  = (tid & 3) * 8;     // A k-col {0,8,16,24}
    const int kB0 = (tid >> 4) * 2;    // B k-row pair base (0..30)
    const int nB  = (tid & 15) * 4;    // B n base (0..60)

    v8f accg[2][2], accu[2][2];
    #pragma unroll
    for (int i = 0; i < 2; ++i)
        #pragma unroll
        for (int j = 0; j < 2; ++j) { accg[i][j] = (v8f)(0.0f); accu[i][j] = (v8f)(0.0f); }

    float4 aR[2][2], gR[2], uR[2];

    auto loadTiles = [&](int k0) {
        #pragma unroll
        for (int q = 0; q < 2; ++q) {
            const float* pa = Ab + (size_t)(rA + 64 * q) * H_ + k0 + kA;
            aR[q][0] = *(const float4*)pa;
            aR[q][1] = *(const float4*)(pa + 4);
        }
        #pragma unroll
        for (int q = 0; q < 2; ++q) {
            gR[q] = *(const float4*)(Gb + (size_t)(k0 + kB0 + q) * D_ + nB);
            uR[q] = *(const float4*)(Ub + (size_t)(k0 + kB0 + q) * D_ + nB);
        }
    };
    auto storeTiles = [&](u16* A, u16* G, u16* U) {
        #pragma unroll
        for (int q = 0; q < 2; ++q) {
            uint4 w;
            w.x = pkbf(aR[q][0].x, aR[q][0].y);
            w.y = pkbf(aR[q][0].z, aR[q][0].w);
            w.z = pkbf(aR[q][1].x, aR[q][1].y);
            w.w = pkbf(aR[q][1].z, aR[q][1].w);
            *(uint4*)&A[(rA + 64 * q) * KP + kA] = w;
        }
        *(u32*)&G[(nB + 0) * KP + kB0] = pkbf(gR[0].x, gR[1].x);
        *(u32*)&G[(nB + 1) * KP + kB0] = pkbf(gR[0].y, gR[1].y);
        *(u32*)&G[(nB + 2) * KP + kB0] = pkbf(gR[0].z, gR[1].z);
        *(u32*)&G[(nB + 3) * KP + kB0] = pkbf(gR[0].w, gR[1].w);
        *(u32*)&U[(nB + 0) * KP + kB0] = pkbf(uR[0].x, uR[1].x);
        *(u32*)&U[(nB + 1) * KP + kB0] = pkbf(uR[0].y, uR[1].y);
        *(u32*)&U[(nB + 2) * KP + kB0] = pkbf(uR[0].z, uR[1].z);
        *(u32*)&U[(nB + 3) * KP + kB0] = pkbf(uR[0].w, uR[1].w);
    };
    auto compute = [&](const u16* A, const u16* G, const u16* U) {
        FragAB a[2], bg[2], bu[2];
        #pragma unroll
        for (int i = 0; i < 2; ++i) {
            const u16* p = &A[(wm + 16 * i + mr) * KP + hf * 8];
            a[i].u[0] = *(const uint4*)p;          // K = hf*8 + 0..7
            a[i].u[1] = *(const uint4*)(p + 16);   // K = 16 + hf*8 + 0..7
        }
        #pragma unroll
        for (int j = 0; j < 2; ++j) {
            const u16* pg = &G[(wn + 16 * j + mr) * KP + hf * 8];
            bg[j].u[0] = *(const uint4*)pg;        // K = hf*8 + 0..7, 16 + hf*8 + 0..7
            bg[j].u[1] = *(const uint4*)(pg + 16);
            const u16* pu = &U[(wn + 16 * j + mr) * KP + hf * 8];
            bu[j].u[0] = *(const uint4*)pu;
            bu[j].u[1] = *(const uint4*)(pu + 16);
        }
        #pragma unroll
        for (int i = 0; i < 2; ++i)
            #pragma unroll
            for (int j = 0; j < 2; ++j) {
                accg[i][j] = __builtin_amdgcn_wmma_f32_16x16x32_bf16(
                    false, a[i].v, false, bg[j].v, (short)0, accg[i][j], false, false);
                accu[i][j] = __builtin_amdgcn_wmma_f32_16x16x32_bf16(
                    false, a[i].v, false, bu[j].v, (short)0, accu[i][j], false, false);
            }
    };

    loadTiles(0);
    storeTiles(ldsA[0], ldsG[0], ldsU[0]);
    __syncthreads();
    int p = 0;
    for (int k0 = 0; k0 < H_; k0 += 32) {
        const bool more = (k0 + 32) < H_;
        if (more) loadTiles(k0 + 32);
        compute(ldsA[p], ldsG[p], ldsU[p]);
        if (more) {
            storeTiles(ldsA[p ^ 1], ldsG[p ^ 1], ldsU[p ^ 1]);
            __syncthreads();
            p ^= 1;
        }
    }

    // SwiGLU epilogue (C/D layout: M = hf*8 + r, N = lane&15)
    #pragma unroll
    for (int i = 0; i < 2; ++i)
        #pragma unroll
        for (int j = 0; j < 2; ++j)
            #pragma unroll
            for (int r = 0; r < 8; ++r) {
                int t = m0 + wm + 16 * i + hf * 8 + r;
                int d = n0 + wn + 16 * j + mr;
                float g = accg[i][j][r];
                float u = accu[i][j][r];
                float s = u * g * __builtin_amdgcn_rcpf(1.0f + __expf(-g));
                __bf16 hb = (__bf16)s;
                fused[(size_t)(e * T_ + t) * D_ + d] = __builtin_bit_cast(u16, hb);
            }
}

// ---------------------------------------------------------------------------
// Kernel 2: out = fused(bf16, T x D) @ down(D x H), fp32 out.
// ---------------------------------------------------------------------------
__global__ __launch_bounds__(256, 2)
void down_kernel(const u16* __restrict__ fusedAct,
                 const float* __restrict__ Wd,
                 float* __restrict__ out)
{
    __shared__ __align__(16) u16 ldsA[2][128 * KP];
    __shared__ __align__(16) u16 ldsB[2][64 * KP];

    const int tid  = threadIdx.x;
    const int lane = tid & 31;
    const int wv   = tid >> 5;
    const int e    = blockIdx.z;
    const int m0   = blockIdx.y * 128;
    const int n0   = blockIdx.x * 64;

    const u16*   Ab = fusedAct + (size_t)(e * T_ + m0) * D_;
    const float* Bb = Wd + (size_t)e * D_ * H_ + n0;

    const int wm = (wv >> 1) * 32;
    const int wn = (wv & 1)  * 32;
    const int mr = lane & 15;
    const int hf = lane >> 4;

    const int rA  = tid >> 2;
    const int kA  = (tid & 3) * 8;
    const int kB0 = (tid >> 4) * 2;
    const int nB  = (tid & 15) * 4;

    v8f acc[2][2];
    #pragma unroll
    for (int i = 0; i < 2; ++i)
        #pragma unroll
        for (int j = 0; j < 2; ++j) acc[i][j] = (v8f)(0.0f);

    uint4  aR[2];
    float4 bR[2];

    auto loadTiles = [&](int k0) {
        #pragma unroll
        for (int q = 0; q < 2; ++q)
            aR[q] = *(const uint4*)(Ab + (size_t)(rA + 64 * q) * D_ + k0 + kA);
        #pragma unroll
        for (int q = 0; q < 2; ++q)
            bR[q] = *(const float4*)(Bb + (size_t)(k0 + kB0 + q) * H_ + nB);
    };
    auto storeTiles = [&](u16* A, u16* B) {
        #pragma unroll
        for (int q = 0; q < 2; ++q)
            *(uint4*)&A[(rA + 64 * q) * KP + kA] = aR[q];
        *(u32*)&B[(nB + 0) * KP + kB0] = pkbf(bR[0].x, bR[1].x);
        *(u32*)&B[(nB + 1) * KP + kB0] = pkbf(bR[0].y, bR[1].y);
        *(u32*)&B[(nB + 2) * KP + kB0] = pkbf(bR[0].z, bR[1].z);
        *(u32*)&B[(nB + 3) * KP + kB0] = pkbf(bR[0].w, bR[1].w);
    };
    auto compute = [&](const u16* A, const u16* B) {
        FragAB a[2], b[2];
        #pragma unroll
        for (int i = 0; i < 2; ++i) {
            const u16* pa = &A[(wm + 16 * i + mr) * KP + hf * 8];
            a[i].u[0] = *(const uint4*)pa;
            a[i].u[1] = *(const uint4*)(pa + 16);
        }
        #pragma unroll
        for (int j = 0; j < 2; ++j) {
            const u16* pb = &B[(wn + 16 * j + mr) * KP + hf * 8];
            b[j].u[0] = *(const uint4*)pb;
            b[j].u[1] = *(const uint4*)(pb + 16);
        }
        #pragma unroll
        for (int i = 0; i < 2; ++i)
            #pragma unroll
            for (int j = 0; j < 2; ++j)
                acc[i][j] = __builtin_amdgcn_wmma_f32_16x16x32_bf16(
                    false, a[i].v, false, b[j].v, (short)0, acc[i][j], false, false);
    };

    loadTiles(0);
    storeTiles(ldsA[0], ldsB[0]);
    __syncthreads();
    int p = 0;
    for (int k0 = 0; k0 < D_; k0 += 32) {
        const bool more = (k0 + 32) < D_;
        if (more) loadTiles(k0 + 32);
        compute(ldsA[p], ldsB[p]);
        if (more) {
            storeTiles(ldsA[p ^ 1], ldsB[p ^ 1]);
            __syncthreads();
            p ^= 1;
        }
    }

    #pragma unroll
    for (int i = 0; i < 2; ++i)
        #pragma unroll
        for (int j = 0; j < 2; ++j)
            #pragma unroll
            for (int r = 0; r < 8; ++r) {
                int t = m0 + wm + 16 * i + hf * 8 + r;
                int h = n0 + wn + 16 * j + mr;
                out[(size_t)(e * T_ + t) * H_ + h] = acc[i][j][r];
            }
}

extern "C" void kernel_launch(void* const* d_in, const int* in_sizes, int n_in,
                              void* d_out, int out_size, void* d_ws, size_t ws_size,
                              hipStream_t stream)
{
    (void)in_sizes; (void)n_in; (void)out_size; (void)ws_size;
    const float* hs   = (const float*)d_in[0];
    const float* gate = (const float*)d_in[1];
    const float* up   = (const float*)d_in[2];
    const float* down = (const float*)d_in[3];
    u16*   fused = (u16*)d_ws;   // E*T*D bf16 = 128 MiB scratch
    float* outp  = (float*)d_out;

    dim3 blk(256);
    gateup_kernel<<<dim3(D_ / 64, T_ / 128, E_), blk, 0, stream>>>(hs, gate, up, fused);
    down_kernel<<<dim3(H_ / 64, T_ / 128, E_), blk, 0, stream>>>(fused, down, outp);
}
